// RelationAwareLayer_81149112091096
// MI455X (gfx1250) — matrix operand order repaired and checked by reading, not compiled, not run on hardware
//
#include <hip/hip_runtime.h>

typedef __attribute__((ext_vector_type(16))) __bf16 v16bf;
typedef __attribute__((ext_vector_type(8)))  float  v8f;
typedef unsigned int u32;
typedef __attribute__((ext_vector_type(4))) u32 u32x4;
typedef __attribute__((ext_vector_type(8))) int i32x8;
typedef __attribute__((ext_vector_type(4))) int i32x4;

#define HEADS  8
#define HEAD_D 16
#define DIM    128
#define NT     4
#define RT     8

// ---------------------------------------------------------------------------
// TDM: DMA a rows x cols f32 tile (row stride in elements) from global to LDS.
// D# layout per CDNA5 ISA:
//   g0: [1:0]=count=1 | [63:32]=lds_addr | [120:64]=global_addr | [127:126]=type=2
//   g1: [17:16]=data_size(2 -> 4B) | [79:48]=tensor_dim0 | [111:80]=tensor_dim1
//       [127:112]=tile_dim0 | [143:128]=tile_dim1 | [207:160]=tensor_dim0_stride
// OOB reads (row >= tensor_dim1) return zero -> free edge-tile zero padding.
// This toolchain exposes the 6-arg builtin:
//   (u32x4 g0, i32x8 g1, i32x4 g2, i32x4 g3, i32x8 g4, i32 cpol)
// ---------------------------------------------------------------------------
__device__ __forceinline__ void tdm_load_tile_f32(u32 lds_off, const float* gptr,
                                                  u32 rows, u32 cols,
                                                  u32 row_stride_elems) {
  unsigned long long ga = (unsigned long long)(uintptr_t)gptr;
  u32x4 g0;
  g0[0] = 1u;                                              // count=1
  g0[1] = lds_off;                                         // lds_addr
  g0[2] = (u32)(ga & 0xFFFFFFFFu);                         // global_addr[31:0]
  g0[3] = (u32)((ga >> 32) & 0x01FFFFFFu) | 0x80000000u;   // addr[56:32] | type=2
  i32x8 g1;
  g1[0] = (int)(2u << 16);                                 // data_size = 4 bytes
  g1[1] = (int)((cols & 0xFFFFu) << 16);                   // tensor_dim0 lo16 (bits 63:48)
  g1[2] = (int)(((cols >> 16) & 0xFFFFu) | ((rows & 0xFFFFu) << 16)); // dim0 hi | dim1 lo
  g1[3] = (int)(((rows >> 16) & 0xFFFFu) | ((cols & 0xFFFFu) << 16)); // dim1 hi | tile_dim0
  g1[4] = (int)(rows & 0xFFFFu);                           // tile_dim1 | tile_dim2=0
  g1[5] = (int)row_stride_elems;                           // tensor_dim0_stride[31:0]
  g1[6] = 0;
  g1[7] = 0;
  i32x4 gz4 = {0, 0, 0, 0};
  i32x8 gz8 = {0, 0, 0, 0, 0, 0, 0, 0};
  __builtin_amdgcn_tensor_load_to_lds(g0, g1, gz4, gz4, gz8, 0);
}

// ---------------------------------------------------------------------------
// WMMA fragment helpers (layouts per CDNA5 ISA 7.12.2, wave32)
// A (16x32 bf16): lanes 0-15 row=lane, K = {0..7,16..23}; lanes 16-31: +8
// B (32x16 bf16): col = lane&15, K = ((lane>=16)?8:0) + j + ((j>=8)?8:0)
// C/D (16x16 f32): elem i -> M = i + ((lane>=16)?8:0), N = lane&15
// ---------------------------------------------------------------------------
__device__ __forceinline__ v16bf load_a_frag_f32(const float* rowp, int k0, int hi) {
  v16bf a;
#pragma unroll
  for (int j = 0; j < 16; ++j) {
    int kk = k0 + j + ((j >= 8) ? 8 : 0) + (hi ? 8 : 0);
    a[j] = (__bf16)rowp[kk];
  }
  return a;
}

// packed-B layout: frag = ((m*NT + t)*4 + ks)*8 + wave ; 32 lanes x 16 bf16
__device__ __forceinline__ v16bf load_b_packed(const __bf16* __restrict__ P,
                                               int m, int t, int ks, int wave, int lane) {
  const __bf16* p = P + ((size_t)((((m * NT + t) * 4 + ks) * 8 + wave)) << 9) + (lane << 4);
  return *(const v16bf*)p;
}

// ---------------------------------------------------------------------------
// 0a) workspace init (amax=-inf, esum=0, agg=0) — must run every launch
// ---------------------------------------------------------------------------
__global__ __launch_bounds__(256) void init_kernel(float* __restrict__ amax,
                                                   float* __restrict__ esum,
                                                   float* __restrict__ agg, int N) {
  size_t i = (size_t)blockIdx.x * 256 + threadIdx.x;
  if (i < (size_t)N * HEADS) { amax[i] = -3.402823466e38f; esum[i] = 0.f; }
  if (i < (size_t)N * DIM)   agg[i] = 0.f;
}

// ---------------------------------------------------------------------------
// 0b) pack Wq/Wk/Wv/Wa (f32 [T,in,out]) into WMMA B-fragment-ordered bf16 so
//     each lane loads its v16bf with one contiguous 32B access.
// ---------------------------------------------------------------------------
__global__ __launch_bounds__(256) void pack_w_kernel(
    const float* __restrict__ Wq, const float* __restrict__ Wk,
    const float* __restrict__ Wv, const float* __restrict__ Wa,
    __bf16* __restrict__ P) {
  int idx = blockIdx.x * 256 + threadIdx.x;          // 4*4*4*8*32*16 = 262144
  if (idx >= 4 * NT * 4 * 8 * 512) return;
  int j    = idx & 15;
  int lane = (idx >> 4) & 31;
  int w    = (idx >> 9) & 7;
  int ks   = (idx >> 12) & 3;
  int t    = (idx >> 14) & 3;
  int m    = idx >> 16;
  const float* W = (m == 0) ? Wq : (m == 1) ? Wk : (m == 2) ? Wv : Wa;
  int kbase = (lane >= 16) ? 8 : 0;
  int K = ks * 32 + kbase + j + ((j >= 8) ? 8 : 0);
  int o = w * 16 + (lane & 15);
  P[idx] = (__bf16)W[(size_t)t * DIM * DIM + (size_t)K * DIM + o];
}

// ---------------------------------------------------------------------------
// 1) typed linear q/k/v via bf16 WMMA with per-row type masking:
//    y = sum_t (mask_t . h_tile) x W_t  -> exact typed_linear, dense WMMA.
//    A tile staged by TDM (tensor_load_to_lds), B from packed bf16 fragments.
// ---------------------------------------------------------------------------
__global__ __launch_bounds__(256) void qkv_wmma_kernel(
    const float* __restrict__ h, const __bf16* __restrict__ P,
    const int* __restrict__ ntype,
    float* __restrict__ q, float* __restrict__ k, float* __restrict__ v, int N) {
  __shared__ struct { float A[16][DIM]; int type[16]; } sm;
  const int row0 = blockIdx.x * 16;
  const int tid = threadIdx.x;
  const u32 rows = (u32)((N - row0 < 16) ? (N - row0) : 16);

  if (tid < 32) {                      // wave 0 issues the tile DMA and waits
    tdm_load_tile_f32((u32)(uintptr_t)&sm.A[0][0], h + (size_t)row0 * DIM,
                      rows, DIM, DIM);
    __builtin_amdgcn_s_wait_tensorcnt(0);
  }
  if (tid >= 32 && tid < 48) {
    int r = tid - 32;
    sm.type[r] = (row0 + r < N) ? ntype[row0 + r] : -1;
  }
  __syncthreads();

  const int wave = tid >> 5;           // 8 waves -> 8 N-tiles of 16 (128 outs)
  const int lane = tid & 31;
  const int r    = lane & 15;          // A row within tile == B/D column
  const int hi   = lane >> 4;
  const int col  = wave * 16 + r;
  const int myType = sm.type[r];

  v8f accq = {}, acck = {}, accv = {};
  const v16bf zero = {};

#pragma unroll
  for (int ks = 0; ks < 4; ++ks) {
    const int k0 = ks * 32;
    v16bf afull = load_a_frag_f32(&sm.A[r][0], k0, hi);
#pragma unroll
    for (int t = 0; t < NT; ++t) {
      v16bf at = (myType == t) ? afull : zero;
      v16bf bq = load_b_packed(P, 0, t, ks, wave, lane);
      v16bf bk = load_b_packed(P, 1, t, ks, wave, lane);
      v16bf bv = load_b_packed(P, 2, t, ks, wave, lane);
      accq = __builtin_amdgcn_wmma_f32_16x16x32_bf16(false, at, false, bq, (short)0, accq, false, false);
      acck = __builtin_amdgcn_wmma_f32_16x16x32_bf16(false, at, false, bk, (short)0, acck, false, false);
      accv = __builtin_amdgcn_wmma_f32_16x16x32_bf16(false, at, false, bv, (short)0, accv, false, false);
    }
  }

#pragma unroll
  for (int i = 0; i < 8; ++i) {
    int rr = row0 + i + hi * 8;
    if (rr < N) {
      size_t o = (size_t)rr * DIM + col;
      q[o] = accq[i]; k[o] = acck[i]; v[o] = accv[i];
    }
  }
}

// ---------------------------------------------------------------------------
// 2) edge scores: a = (ke W_att[h,r]) . qe * pri / sqrt(16); atomic seg-max.
//    rel_att (64KB) staged whole into LDS; k/q gathers as float4.
// ---------------------------------------------------------------------------
__global__ __launch_bounds__(256) void edge_att_kernel(
    const float* __restrict__ q, const float* __restrict__ k,
    const float* __restrict__ rel_att, const float* __restrict__ rel_pri,
    const int* __restrict__ etype, const int* __restrict__ src,
    const int* __restrict__ dst, float* __restrict__ a,
    float* __restrict__ amax, int E) {
  __shared__ float sW[HEADS * RT * 256];     // 64 KB
  const int tid = threadIdx.x;
  for (int i = tid; i < HEADS * RT * 256; i += 256) sW[i] = rel_att[i];
  __syncthreads();

  size_t idx = (size_t)blockIdx.x * 256 + tid;
  if (idx >= (size_t)E * HEADS) return;
  int e = (int)(idx >> 3), hh = (int)(idx & 7);
  int rel = etype[e];
  const float4* ke4 = (const float4*)(k + (size_t)src[e] * DIM + hh * HEAD_D);
  const float4* qe4 = (const float4*)(q + (size_t)dst[e] * DIM + hh * HEAD_D);
  float kv[16], qv[16];
#pragma unroll
  for (int i = 0; i < 4; ++i) {
    ((float4*)kv)[i] = ke4[i];
    ((float4*)qv)[i] = qe4[i];
  }
  const float* W = &sW[(hh * RT + rel) << 8];
  float acc = 0.f;
#pragma unroll
  for (int o = 0; o < 16; ++o) {
    float s = 0.f;
#pragma unroll
    for (int d = 0; d < 16; ++d) s = fmaf(kv[d], W[d * 16 + o], s);
    acc = fmaf(s, qv[o], acc);
  }
  acc *= rel_pri[hh * RT + rel] * 0.25f;     // 1/sqrt(HEAD_D)
  a[idx] = acc;
  float* am = amax + (size_t)dst[e] * HEADS + hh;
  if (acc >= 0.f) atomicMax((int*)am, __float_as_int(acc));
  else            atomicMin((unsigned int*)am, __float_as_uint(acc));
}

// ---------------------------------------------------------------------------
// 3) exp(a - amax[dst]) and atomic segment-sum
// ---------------------------------------------------------------------------
__global__ __launch_bounds__(256) void edge_exp_kernel(
    float* __restrict__ a, const float* __restrict__ amax,
    float* __restrict__ esum, const int* __restrict__ dst, int E) {
  size_t idx = (size_t)blockIdx.x * 256 + threadIdx.x;
  if (idx >= (size_t)E * HEADS) return;
  int e = (int)(idx >> 3), hh = (int)(idx & 7);
  size_t nh = (size_t)dst[e] * HEADS + hh;
  float ea = __expf(a[idx] - amax[nh]);
  a[idx] = ea;
  atomicAdd(&esum[nh], ea);
}

// ---------------------------------------------------------------------------
// 4) relation message m = ve W_msg[h,r]; scatter-add attn*m into agg[dst].
//    rel_msg staged whole into LDS.
// ---------------------------------------------------------------------------
__global__ __launch_bounds__(256) void edge_agg_kernel(
    const float* __restrict__ a, const float* __restrict__ esum,
    const float* __restrict__ v, const float* __restrict__ rel_msg,
    const int* __restrict__ etype, const int* __restrict__ src,
    const int* __restrict__ dst, float* __restrict__ agg, int E) {
  __shared__ float sW[HEADS * RT * 256];     // 64 KB
  const int tid = threadIdx.x;
  for (int i = tid; i < HEADS * RT * 256; i += 256) sW[i] = rel_msg[i];
  __syncthreads();

  size_t idx = (size_t)blockIdx.x * 256 + tid;
  if (idx >= (size_t)E * HEADS) return;
  int e = (int)(idx >> 3), hh = (int)(idx & 7);
  int rel = etype[e];
  int dd = dst[e];
  float attn = a[idx] / esum[(size_t)dd * HEADS + hh];
  const float4* ve4 = (const float4*)(v + (size_t)src[e] * DIM + hh * HEAD_D);
  float vv[16];
#pragma unroll
  for (int i = 0; i < 4; ++i) ((float4*)vv)[i] = ve4[i];
  const float* W = &sW[(hh * RT + rel) << 8];
  float* outp = agg + (size_t)dd * DIM + hh * HEAD_D;
#pragma unroll
  for (int o = 0; o < 16; ++o) {
    float s = 0.f;
#pragma unroll
    for (int d = 0; d < 16; ++d) s = fmaf(vv[d], W[d * 16 + o], s);
    atomicAdd(&outp[o], s * attn);
  }
}

// ---------------------------------------------------------------------------
// 5) output projection (typed WMMA) + gated skip + residual, in place on agg:
//    x = ha*alpha + h*(2 - alpha). A tile via TDM, B from packed fragments.
// ---------------------------------------------------------------------------
__global__ __launch_bounds__(256) void out_wmma_kernel(
    const float* agg, const float* __restrict__ h,
    const __bf16* __restrict__ P, const float* __restrict__ skip,
    const int* __restrict__ ntype, float* x, int N) {
  __shared__ struct { float A[16][DIM]; int type[16]; float alpha[16]; } sm;
  const int row0 = blockIdx.x * 16;
  const int tid = threadIdx.x;
  const u32 rows = (u32)((N - row0 < 16) ? (N - row0) : 16);

  if (tid < 32) {
    tdm_load_tile_f32((u32)(uintptr_t)&sm.A[0][0], agg + (size_t)row0 * DIM,
                      rows, DIM, DIM);
    __builtin_amdgcn_s_wait_tensorcnt(0);
  }
  if (tid >= 32 && tid < 48) {
    int r = tid - 32;
    int t = (row0 + r < N) ? ntype[row0 + r] : -1;
    sm.type[r] = t;
    sm.alpha[r] = (t >= 0) ? 1.f / (1.f + __expf(-skip[t])) : 0.f;
  }
  __syncthreads();

  const int wave = tid >> 5;
  const int lane = tid & 31;
  const int r    = lane & 15;
  const int hi   = lane >> 4;
  const int col  = wave * 16 + r;
  const int myType = sm.type[r];

  v8f acc = {};
  const v16bf zero = {};

#pragma unroll
  for (int ks = 0; ks < 4; ++ks) {
    const int k0 = ks * 32;
    v16bf afull = load_a_frag_f32(&sm.A[r][0], k0, hi);
#pragma unroll
    for (int t = 0; t < NT; ++t) {
      v16bf at = (myType == t) ? afull : zero;
      v16bf ba = load_b_packed(P, 3, t, ks, wave, lane);
      acc = __builtin_amdgcn_wmma_f32_16x16x32_bf16(false, at, false, ba, (short)0, acc, false, false);
    }
  }

#pragma unroll
  for (int i = 0; i < 8; ++i) {
    int rr = row0 + i + hi * 8;
    if (rr < N) {
      float al = sm.alpha[i + hi * 8];
      size_t o = (size_t)rr * DIM + col;
      x[o] = acc[i] * al + h[o] * (2.f - al);
    }
  }
}

// ---------------------------------------------------------------------------
// 6) row LayerNorm: one wave32 per node, 4 elems/lane, shfl_xor reduction
// ---------------------------------------------------------------------------
__global__ __launch_bounds__(256) void ln_kernel(
    const float* __restrict__ x, const float* __restrict__ gamma,
    const float* __restrict__ beta, float* __restrict__ out, int N) {
  int node = (int)(((size_t)blockIdx.x * 256 + threadIdx.x) >> 5);
  int lane = threadIdx.x & 31;
  if (node >= N) return;
  const float* xr = x + (size_t)node * DIM;
  float vals[4], s = 0.f, ss = 0.f;
#pragma unroll
  for (int i = 0; i < 4; ++i) {
    float f = xr[lane + i * 32];
    vals[i] = f; s += f; ss = fmaf(f, f, ss);
  }
#pragma unroll
  for (int off = 16; off >= 1; off >>= 1) {
    s  += __shfl_xor(s, off, 32);
    ss += __shfl_xor(ss, off, 32);
  }
  float mu  = s * (1.f / DIM);
  float var = ss * (1.f / DIM) - mu * mu;
  float inv = rsqrtf(var + 1e-5f);
#pragma unroll
  for (int i = 0; i < 4; ++i) {
    int c = lane + i * 32;
    out[(size_t)node * DIM + c] = (vals[i] - mu) * inv * gamma[c] + beta[c];
  }
}

// ---------------------------------------------------------------------------
extern "C" void kernel_launch(void* const* d_in, const int* in_sizes, int n_in,
                              void* d_out, int out_size, void* d_ws, size_t ws_size,
                              hipStream_t stream) {
  const float* h       = (const float*)d_in[0];
  const float* Wk      = (const float*)d_in[1];
  const float* Wq      = (const float*)d_in[2];
  const float* Wv      = (const float*)d_in[3];
  const float* Wa      = (const float*)d_in[4];
  const float* rel_att = (const float*)d_in[5];
  const float* rel_msg = (const float*)d_in[6];
  const float* rel_pri = (const float*)d_in[7];
  const float* skip    = (const float*)d_in[8];
  const float* gamma   = (const float*)d_in[9];
  const float* beta    = (const float*)d_in[10];
  const int* ntype = (const int*)d_in[11];
  const int* etype = (const int*)d_in[12];
  const int* src   = (const int*)d_in[13];
  const int* dst   = (const int*)d_in[14];
  const int N = in_sizes[11];
  const int E = in_sizes[12];

  // workspace: q,k,v [N*128] | a [E*8] | amax,esum [N*8] | agg [N*128] | packed W (bf16)
  float* ws   = (float*)d_ws;
  float* q    = ws;
  float* k    = q    + (size_t)N * DIM;
  float* v    = k    + (size_t)N * DIM;
  float* a    = v    + (size_t)N * DIM;
  float* amax = a    + (size_t)E * HEADS;
  float* esum = amax + (size_t)N * HEADS;
  float* agg  = esum + (size_t)N * HEADS;
  __bf16* P   = (__bf16*)(agg + (size_t)N * DIM);   // 4*4*4*8*512 bf16 = 512KB

  const unsigned ib = (unsigned)(((size_t)N * DIM + 255) / 256);
  const unsigned nb = (unsigned)((N + 15) / 16);
  const unsigned eb = (unsigned)(((size_t)E * HEADS + 255) / 256);
  const unsigned lb = (unsigned)((N + 7) / 8);

  init_kernel<<<ib, 256, 0, stream>>>(amax, esum, agg, N);
  pack_w_kernel<<<1024, 256, 0, stream>>>(Wq, Wk, Wv, Wa, P);
  qkv_wmma_kernel<<<nb, 256, 0, stream>>>(h, P, ntype, q, k, v, N);
  edge_att_kernel<<<eb, 256, 0, stream>>>(q, k, rel_att, rel_pri, etype, src, dst, a, amax, E);
  edge_exp_kernel<<<eb, 256, 0, stream>>>(a, amax, esum, dst, E);
  edge_agg_kernel<<<eb, 256, 0, stream>>>(a, esum, v, rel_msg, etype, src, dst, agg, E);
  out_wmma_kernel<<<nb, 256, 0, stream>>>(agg, h, P, skip, ntype, agg, N);
  ln_kernel<<<lb, 256, 0, stream>>>(agg, gamma, beta, (float*)d_out, N);
}
